// TinySwinTransformer_38096359915902
// MI455X (gfx1250) — matrix operand order repaired and checked by reading, not compiled, not run on hardware
//
#include <hip/hip_runtime.h>
#include <math.h>

// ---------------------------------------------------------------------------
// MI455X (gfx1250) TinySwin forward.
// Memory-bound ends (~800MB HBM traffic ~= 35us @ 23.3TB/s) are fused single
// pass kernels. The ~29 GFLOP transformer core runs on v_wmma_f32_16x16x32_f16
// with flash-attention (no 4096^2 score materialization, rel-pos bias computed
// on the fly). GEMM uses 2x2 register tiles (4 WMMA / K-step). Attention
// stages K/V chunks into LDS via global_load_async_to_lds_b128 (ASYNCcnt,
// double buffered) when the toolchain exposes it.
// ---------------------------------------------------------------------------

typedef __attribute__((ext_vector_type(16))) _Float16 v16h;
typedef __attribute__((ext_vector_type(8)))  _Float16 v8h;
typedef __attribute__((ext_vector_type(8)))  float    v8f;
typedef __attribute__((ext_vector_type(4)))  int      v4i;

#define NTOK 4096     // 64*64 tokens
#define EDIM 96

#if __has_builtin(__builtin_amdgcn_global_load_async_to_lds_b128) && \
    __has_builtin(__builtin_amdgcn_s_wait_asynccnt)
#define HAVE_ASYNC_LDS 1
#else
#define HAVE_ASYNC_LDS 0
#endif

#if HAVE_ASYNC_LDS
// ROCm clang-22 signature (from diagnostics): params are pointers to 16-byte
// int vectors (generic AS), then immediate offset + cpol.
__device__ __forceinline__ void async_cp16(const void* g, void* l) {
  __builtin_amdgcn_global_load_async_to_lds_b128((v4i*)g, (v4i*)l, 0, 0);
}
#endif

// ---------------------------------------------------------------------------
// Stage 1: depth-mean over D=16 fused with 2x2 avg (bilinear /2, align=False
// degenerates to exact 2x2 average). Thread id = c*4096 + pixel for coalesced
// reads (64 reads per 1 write). Output pix[pixel][channel] (GEMM A layout).
// ---------------------------------------------------------------------------
__global__ void k_mean_downsample(const float* __restrict__ x,
                                  float* __restrict__ pix) {
  int idx = blockIdx.x * blockDim.x + threadIdx.x;
  if (idx >= 256 * NTOK) return;
  int c = idx >> 12;
  int p = idx & 4095;
  int oy = p >> 6, ox = p & 63;
  float s = 0.f;
#pragma unroll 4
  for (int d = 0; d < 16; ++d) {
    const float2* row0 =
        (const float2*)(x + ((size_t)(c * 16 + d) * 128 + 2 * oy) * 128) + ox;
    const float2* row1 = row0 + 64;  // next image row
    float2 a = *row0, b = *row1;
    s += a.x + a.y + b.x + b.y;
  }
  pix[(size_t)p * 256 + c] = s * (1.f / 64.f);
}

// ---------------------------------------------------------------------------
// Fold pre_conv (256->64) and patch embed (64->96) into one 96x256 weight.
// ---------------------------------------------------------------------------
__global__ void k_wcomb(const float* __restrict__ patch_w,
                        const float* __restrict__ patch_b,
                        const float* __restrict__ pre_w,
                        const float* __restrict__ pre_b,
                        float* __restrict__ Wc, float* __restrict__ bc) {
  int idx = blockIdx.x * blockDim.x + threadIdx.x;
  if (idx < 96 * 256) {
    int o = idx >> 8, c = idx & 255;
    float s = 0.f;
#pragma unroll 8
    for (int k = 0; k < 64; ++k) s += patch_w[o * 64 + k] * pre_w[k * 256 + c];
    Wc[(size_t)o * 256 + c] = s;
  }
  if (idx < 96) {
    float s = patch_b[idx];
#pragma unroll 8
    for (int k = 0; k < 64; ++k) s += patch_w[idx * 64 + k] * pre_b[k];
    bc[idx] = s;
  }
}

// ---- WMMA fragment loaders (CDNA5 16-bit A 16x32 / B 32x16 layouts) --------
// A: lane&15 = row M; elems 0..7 -> K = hi*8+e ; elems 8..15 -> K = 16+hi*8+e
// B: lane&15 = col N; elems 0..7 -> K = hi*8+e ; elems 8..15 -> K = 16+hi*8+e
__device__ __forceinline__ v16h load16_a_f32(const float* p) {
  float4 u0 = *(const float4*)(p);
  float4 u1 = *(const float4*)(p + 4);
  float4 u2 = *(const float4*)(p + 16);
  float4 u3 = *(const float4*)(p + 20);
  v16h a;
  a[0] = (_Float16)u0.x;  a[1] = (_Float16)u0.y;
  a[2] = (_Float16)u0.z;  a[3] = (_Float16)u0.w;
  a[4] = (_Float16)u1.x;  a[5] = (_Float16)u1.y;
  a[6] = (_Float16)u1.z;  a[7] = (_Float16)u1.w;
  a[8] = (_Float16)u2.x;  a[9] = (_Float16)u2.y;
  a[10] = (_Float16)u2.z; a[11] = (_Float16)u2.w;
  a[12] = (_Float16)u3.x; a[13] = (_Float16)u3.y;
  a[14] = (_Float16)u3.z; a[15] = (_Float16)u3.w;
  return a;
}
__device__ __forceinline__ v16h load16_b_f32(const float* p) {
  float4 u0 = *(const float4*)(p);
  float4 u1 = *(const float4*)(p + 4);
  float4 u2 = *(const float4*)(p + 16);
  float4 u3 = *(const float4*)(p + 20);
  v16h b;
  b[0] = (_Float16)u0.x;  b[1] = (_Float16)u0.y;
  b[2] = (_Float16)u0.z;  b[3] = (_Float16)u0.w;
  b[4] = (_Float16)u1.x;  b[5] = (_Float16)u1.y;
  b[6] = (_Float16)u1.z;  b[7] = (_Float16)u1.w;
  b[8] = (_Float16)u2.x;  b[9] = (_Float16)u2.y;
  b[10] = (_Float16)u2.z; b[11] = (_Float16)u2.w;
  b[12] = (_Float16)u3.x; b[13] = (_Float16)u3.y;
  b[14] = (_Float16)u3.z; b[15] = (_Float16)u3.w;
  return b;
}

// ---------------------------------------------------------------------------
// Generic C = act(A @ W^T + bias) [+ R]. A: MxK f32, W: NxK f32 row-major.
// One wave per 32x32 C tile: 2x2 register tiling -> 4 WMMAs per K-step,
// halves A and B fetch traffic vs 16x16 tiles. K multiple of 32.
// ---------------------------------------------------------------------------
__global__ void __launch_bounds__(128) k_gemm(
    const float* __restrict__ A, const float* __restrict__ W,
    const float* __restrict__ bias, const float* __restrict__ R,
    float* __restrict__ C, int M, int N, int K, int act) {
  int wid = (blockIdx.x * blockDim.x + threadIdx.x) >> 5;
  int lane = threadIdx.x & 31;
  int lo = lane & 15, hi = lane >> 4;
  int n2 = N >> 5;
  int mt = wid / n2, nt = wid - mt * n2;
  if (mt >= (M >> 5)) return;
  int row = mt << 5;
  int col = (nt << 5) + lo;

  const float* a0p = A + (size_t)(row + lo) * K + hi * 8;
  const float* a1p = a0p + (size_t)16 * K;
  const float* w0p = W + (size_t)col * K + hi * 8;
  const float* w1p = w0p + (size_t)16 * K;

  v8f acc00 = {}, acc01 = {}, acc10 = {}, acc11 = {};
  for (int k0 = 0; k0 < K; k0 += 32) {
    v16h a0 = load16_a_f32(a0p + k0);
    v16h a1 = load16_a_f32(a1p + k0);
    v16h b0 = load16_b_f32(w0p + k0);
    v16h b1 = load16_b_f32(w1p + k0);
    acc00 = __builtin_amdgcn_wmma_f32_16x16x32_f16(false, a0, false, b0,
                                                   (short)0, acc00, false, false);
    acc01 = __builtin_amdgcn_wmma_f32_16x16x32_f16(false, a0, false, b1,
                                                   (short)0, acc01, false, false);
    acc10 = __builtin_amdgcn_wmma_f32_16x16x32_f16(false, a1, false, b0,
                                                   (short)0, acc10, false, false);
    acc11 = __builtin_amdgcn_wmma_f32_16x16x32_f16(false, a1, false, b1,
                                                   (short)0, acc11, false, false);
  }

  float b0c = bias[col], b1c = bias[col + 16];
#define EPI(vv, rr, cc, bb)                                              \
  {                                                                      \
    float v = (vv) + (bb);                                               \
    if (act == 1) v = 0.5f * v * (1.f + erff(v * 0.70710678118654752f)); \
    if (R) v += R[(size_t)(rr) * N + (cc)];                              \
    C[(size_t)(rr) * N + (cc)] = v;                                      \
  }
#pragma unroll
  for (int r = 0; r < 8; ++r) {
    int r0 = row + hi * 8 + r;
    int r1 = r0 + 16;
    EPI(acc00[r], r0, col, b0c);
    EPI(acc01[r], r0, col + 16, b1c);
    EPI(acc10[r], r1, col, b0c);
    EPI(acc11[r], r1, col + 16, b1c);
  }
#undef EPI
}

// ---------------------------------------------------------------------------
// LayerNorm over E=96, one wave per row (3 elems/lane, shfl_xor reductions).
// ---------------------------------------------------------------------------
__global__ void k_layernorm(const float* __restrict__ X,
                            const float* __restrict__ g,
                            const float* __restrict__ b,
                            float* __restrict__ Y, int Mrows) {
  int wid = (blockIdx.x * blockDim.x + threadIdx.x) >> 5;
  int lane = threadIdx.x & 31;
  if (wid >= Mrows) return;
  const float* xr = X + (size_t)wid * EDIM;
  float v0 = xr[lane], v1 = xr[lane + 32], v2 = xr[lane + 64];
  float s = v0 + v1 + v2;
#pragma unroll
  for (int m = 16; m >= 1; m >>= 1) s += __shfl_xor(s, m);
  float mean = s * (1.f / 96.f);
  float d0 = v0 - mean, d1 = v1 - mean, d2 = v2 - mean;
  float q = d0 * d0 + d1 * d1 + d2 * d2;
#pragma unroll
  for (int m = 16; m >= 1; m >>= 1) q += __shfl_xor(q, m);
  float inv = rsqrtf(q * (1.f / 96.f) + 1e-5f);
  float* yr = Y + (size_t)wid * EDIM;
  yr[lane]      = d0 * inv * g[lane]      + b[lane];
  yr[lane + 32] = d1 * inv * g[lane + 32] + b[lane + 32];
  yr[lane + 64] = d2 * inv * g[lane + 64] + b[lane + 64];
}

// ---------------------------------------------------------------------------
// Split QKV (4096x288 f32) into f16 Q(scaled)/K: [h][tok][dh] and V
// *transposed*: vt[h][d][tok] so P@V B-fragments are single v16h loads.
// ---------------------------------------------------------------------------
__global__ void k_repack_qkv(const float* __restrict__ qkv,
                             _Float16* __restrict__ qh,
                             _Float16* __restrict__ kh,
                             _Float16* __restrict__ vt,
                             int nH, int dh, float qscale) {
  int idx = blockIdx.x * blockDim.x + threadIdx.x;
  if (idx >= NTOK * 288) return;
  int i = idx / 288, r = idx - i * 288;
  int s = r / 96, e = r - s * 96;
  int h = e / dh, d = e - h * dh;
  float v = qkv[idx];
  if (s == 0)      qh[((size_t)h * NTOK + i) * dh + d] = (_Float16)(v * qscale);
  else if (s == 1) kh[((size_t)h * NTOK + i) * dh + d] = (_Float16)v;
  else             vt[((size_t)h * dh + d) * NTOK + i] = (_Float16)v;
}

// ---------------------------------------------------------------------------
// Flash attention, one wave per (head, 16-row Q tile); the 4 waves of a block
// share one head's K/V stream. 32 keys per step: 2x QK^T WMMA -> rel-pos bias
// (computed from (dy,dx); rpb table is L2-resident) -> online softmax
// (shfl_xor within 16-lane groups) -> P through a private 1KB LDS slice
// (same-wave DS ops are in-order: no barrier) -> PV WMMA(s).
// When available, K/V chunks are staged once per block into LDS with
// global_load_async_to_lds_b128 (double buffered, s_wait_asynccnt + barrier).
// ---------------------------------------------------------------------------
__global__ void __launch_bounds__(128) k_attn(
    const _Float16* __restrict__ qh, const _Float16* __restrict__ kh,
    const _Float16* __restrict__ vt, const float* __restrict__ rpb,
    float* __restrict__ out, int nH, int dh) {
  __shared__ _Float16 ldsP[4][16 * 32];
#if HAVE_ASYNC_LDS
  __shared__ _Float16 kstage[2][32 * 32];
  __shared__ _Float16 vstage[2][32 * 32];
#endif
  int widb = threadIdx.x >> 5;
  int wid = blockIdx.x * 4 + widb;
  int lane = threadIdx.x & 31;
  int lo = lane & 15, hi = lane >> 4;
  int h = wid >> 8;  // 256 Q tiles per head; grid sized so h < nH always
  int mt = wid & 255;
  int qrow0 = mt << 4;

  const _Float16* qb = qh + (size_t)h * NTOK * dh;
  const _Float16* kb = kh + (size_t)h * NTOK * dh;
  const _Float16* vb = vt + (size_t)h * dh * NTOK;

  // Q A-fragment (zero-pad K>=16 when dh==16 so padded B lanes multiply by 0)
  v16h aq;
  {
    const _Float16* qr = qb + (size_t)(qrow0 + lo) * dh;
    v8h l0 = {};
    v8h l1 = {};
    if (dh == 32 || hi == 0) {
      l0 = *(const v8h*)(qr + hi * 16);
      l1 = *(const v8h*)(qr + hi * 16 + 8);
    }
    aq = __builtin_shufflevector(l0, l1, 0, 1, 2, 3, 4, 5, 6, 7, 8, 9, 10, 11,
                                 12, 13, 14, 15);
  }

  float m[8], l[8];
  v8f o0 = {}, o1 = {};
  int iy[8], ix[8];
#pragma unroll
  for (int r = 0; r < 8; ++r) {
    m[r] = -1e30f;
    l[r] = 0.f;
    int i = qrow0 + hi * 8 + r;
    iy[r] = i >> 6;
    ix[r] = i & 63;
  }
  _Float16* myP = ldsP[widb];

#if HAVE_ASYNC_LDS
  {  // zero stage buffers once: dh==16 padding reads must stay finite
    int tid0 = threadIdx.x;
    for (int i = tid0; i < 2 * 32 * 32; i += 128) {
      (&kstage[0][0])[i] = (_Float16)0;
      (&vstage[0][0])[i] = (_Float16)0;
    }
  }
  __syncthreads();
  int tid = threadIdx.x;
  int npieces = (64 * dh) >> 4;  // 16B pieces per K (and per V) chunk
  auto stage = [&](int p, int j0) {
    if (tid < npieces) {
      // K chunk: 32 consecutive rows -> one contiguous 64*dh byte block
      async_cp16((const char*)(kb + (size_t)j0 * dh) + tid * 16,
                 (char*)(&kstage[p][0]) + tid * 16);
      // V^T chunk: dh rows x 32 tokens (64B per row, stride NTOK)
      int rowd = tid >> 2, c4 = tid & 3;
      async_cp16((const void*)(vb + (size_t)rowd * NTOK + j0 + c4 * 8),
                 (char*)(&vstage[p][0]) + tid * 16);
    }
  };
  stage(0, 0);
#endif

  for (int j0 = 0; j0 < NTOK; j0 += 32) {
#if HAVE_ASYNC_LDS
    int p = (j0 >> 5) & 1;
    __builtin_amdgcn_s_wait_asynccnt(0);
    __syncthreads();  // chunk p ready for all waves; p^1 fully consumed
    if (j0 + 32 < NTOK) stage(p ^ 1, j0 + 32);
    v16h bk0 = *(const v16h*)(&kstage[p][0] + lo * dh + hi * 16);
    v16h bk1 = *(const v16h*)(&kstage[p][0] + (16 + lo) * dh + hi * 16);
#else
    if (j0 + 32 < NTOK) {  // gfx1250 global_prefetch_b8 on next chunk
      __builtin_prefetch(kb + (size_t)(j0 + 32 + lane) * dh, 0, 3);
      __builtin_prefetch(vb + (size_t)(lane & (dh - 1)) * NTOK + j0 + 32, 0, 3);
    }
    v16h bk0 = *(const v16h*)(kb + (size_t)(j0 + lo) * dh + hi * 16);
    v16h bk1 = *(const v16h*)(kb + (size_t)(j0 + 16 + lo) * dh + hi * 16);
#endif

    v8f s0 = {}, s1 = {};
    s0 = __builtin_amdgcn_wmma_f32_16x16x32_f16(false, aq, false, bk0,
                                                (short)0, s0, false, false);
    s1 = __builtin_amdgcn_wmma_f32_16x16x32_f16(false, aq, false, bk1,
                                                (short)0, s1, false, false);

    int jA = j0 + lo, jB = j0 + 16 + lo;
    int jAy = jA >> 6, jAx = jA & 63, jBy = jB >> 6, jBx = jB & 63;
    float mt8[8];
#pragma unroll
    for (int r = 0; r < 8; ++r) {
      float bA = rpb[((iy[r] - jAy + 63) * 127 + (ix[r] - jAx + 63)) * nH + h];
      float bB = rpb[((iy[r] - jBy + 63) * 127 + (ix[r] - jBx + 63)) * nH + h];
      s0[r] += bA;
      s1[r] += bB;
      float mr = fmaxf(s0[r], s1[r]);
#pragma unroll
      for (int msk = 8; msk >= 1; msk >>= 1) mr = fmaxf(mr, __shfl_xor(mr, msk));
      mt8[r] = mr;
    }
#pragma unroll
    for (int r = 0; r < 8; ++r) {
      float mn = fmaxf(m[r], mt8[r]);
      float alpha = __expf(m[r] - mn);
      m[r] = mn;
      float p0 = __expf(s0[r] - mn);
      float p1 = __expf(s1[r] - mn);
      float rs = p0 + p1;
#pragma unroll
      for (int msk = 8; msk >= 1; msk >>= 1) rs += __shfl_xor(rs, msk);
      l[r] = l[r] * alpha + rs;
      o0[r] *= alpha;
      o1[r] *= alpha;
      myP[(hi * 8 + r) * 32 + lo] = (_Float16)p0;
      myP[(hi * 8 + r) * 32 + 16 + lo] = (_Float16)p1;
    }
    // C-layout -> A-layout via LDS (intra-wave, DS ops execute in order)
    v8h p0v = *(const v8h*)(myP + lo * 32 + hi * 16);
    v8h p1v = *(const v8h*)(myP + lo * 32 + hi * 16 + 8);
    v16h ap = __builtin_shufflevector(p0v, p1v, 0, 1, 2, 3, 4, 5, 6, 7, 8, 9,
                                      10, 11, 12, 13, 14, 15);
    // V^T B-fragments: contiguous v16h (LDS-staged or direct from vt)
#if HAVE_ASYNC_LDS
    v16h bv0 = *(const v16h*)(&vstage[p][0] + lo * 32 + hi * 16);
    v16h bv1 = *(const v16h*)(&vstage[p][0] + (16 + lo) * 32 + hi * 16);
#else
    v16h bv0 = *(const v16h*)(vb + (size_t)lo * NTOK + j0 + hi * 16);
    v16h bv1;
    if (dh == 32)
      bv1 = *(const v16h*)(vb + (size_t)(16 + lo) * NTOK + j0 + hi * 16);
    else
      bv1 = (v16h){};
#endif
    o0 = __builtin_amdgcn_wmma_f32_16x16x32_f16(false, ap, false, bv0,
                                                (short)0, o0, false, false);
    if (dh == 32)
      o1 = __builtin_amdgcn_wmma_f32_16x16x32_f16(false, ap, false, bv1,
                                                  (short)0, o1, false, false);
  }

  int c0 = h * dh;
#pragma unroll
  for (int r = 0; r < 8; ++r) {
    int orow = qrow0 + hi * 8 + r;
    float invl = 1.f / l[r];
    out[(size_t)orow * EDIM + c0 + lo] = o0[r] * invl;
    if (dh == 32) out[(size_t)orow * EDIM + c0 + 16 + lo] = o1[r] * invl;
  }
}

// ---------------------------------------------------------------------------
// Final: bilinear x2 upsample of feats (64x64x256, pixel-major) + sigmoid,
// then out = x * gate broadcast over D=16. Dominant HBM stage (~540MB).
// ---------------------------------------------------------------------------
__global__ void k_final(const float* __restrict__ x,
                        const float* __restrict__ feats,
                        float* __restrict__ out) {
  int idx = blockIdx.x * blockDim.x + threadIdx.x;
  if (idx >= 256 * 128 * 128) return;
  int c = idx >> 14;
  int y = (idx >> 7) & 127;
  int xc = idx & 127;
  float cy = fminf(fmaxf(y * 0.5f - 0.25f, 0.f), 63.f);
  int y0 = (int)cy; int y1 = min(y0 + 1, 63); float wy = cy - (float)y0;
  float cx = fminf(fmaxf(xc * 0.5f - 0.25f, 0.f), 63.f);
  int x0 = (int)cx; int x1 = min(x0 + 1, 63); float wx = cx - (float)x0;
  float f00 = feats[(size_t)((y0 << 6) + x0) * 256 + c];
  float f01 = feats[(size_t)((y0 << 6) + x1) * 256 + c];
  float f10 = feats[(size_t)((y1 << 6) + x0) * 256 + c];
  float f11 = feats[(size_t)((y1 << 6) + x1) * 256 + c];
  float f = (f00 * (1.f - wx) + f01 * wx) * (1.f - wy) +
            (f10 * (1.f - wx) + f11 * wx) * wy;
  float sg = 1.f / (1.f + __expf(-f));
  size_t base = (size_t)c * 16 * 16384 + (size_t)y * 128 + xc;
#pragma unroll
  for (int d = 0; d < 16; ++d) {
    size_t off = base + (size_t)d * 16384;
    out[off] = x[off] * sg;
  }
}

// ---------------------------------------------------------------------------
extern "C" void kernel_launch(void* const* d_in, const int* in_sizes, int n_in,
                              void* d_out, int out_size, void* d_ws,
                              size_t ws_size, hipStream_t stream) {
  (void)n_in; (void)out_size; (void)ws_size;
  auto F = [&](int i) -> const float* { return (const float*)d_in[i]; };

  const float *x, *pre_w, *pre_b, *patch_w, *patch_b, *patch_ng, *patch_nb;
  const float *norm_g, *norm_b, *map_w, *map_b;
  struct Blk {
    const float *n1g, *n1b, *qkv_w, *qkv_b, *proj_w, *proj_b, *rpb;
    const float *n2g, *n2b, *fc1_w, *fc1_b, *fc2_w, *fc2_b;
  } blk[4];
  const int nHs[4] = {3, 3, 6, 6};

  if (in_sizes[0] == 67108864) {
    // dict insertion order: x first, then params as written in the reference
    x = F(0); pre_w = F(1); pre_b = F(2);
    patch_w = F(3); patch_b = F(4); patch_ng = F(5); patch_nb = F(6);
    int p = 7;
    for (int b = 0; b < 4; ++b) {
      blk[b].n1g = F(p++); blk[b].n1b = F(p++);
      blk[b].qkv_w = F(p++); blk[b].qkv_b = F(p++);
      blk[b].proj_w = F(p++); blk[b].proj_b = F(p++);
      blk[b].rpb = F(p++);
      blk[b].n2g = F(p++); blk[b].n2b = F(p++);
      blk[b].fc1_w = F(p++); blk[b].fc1_b = F(p++);
      blk[b].fc2_w = F(p++); blk[b].fc2_b = F(p++);
    }
    norm_g = F(p++); norm_b = F(p++); map_w = F(p++); map_b = F(p++);
  } else {
    // jax pytree (alphabetical) order: params first ('params' < 'x')
    int p = 0;
    for (int b = 0; b < 4; ++b) {
      blk[b].fc1_b = F(p++); blk[b].fc1_w = F(p++);
      blk[b].fc2_b = F(p++); blk[b].fc2_w = F(p++);
      blk[b].n1b = F(p++); blk[b].n1g = F(p++);
      blk[b].n2b = F(p++); blk[b].n2g = F(p++);
      blk[b].proj_b = F(p++); blk[b].proj_w = F(p++);
      blk[b].qkv_b = F(p++); blk[b].qkv_w = F(p++);
      blk[b].rpb = F(p++);
    }
    map_b = F(p++); map_w = F(p++);
    norm_b = F(p++); norm_g = F(p++);
    patch_b = F(p++); patch_nb = F(p++); patch_ng = F(p++); patch_w = F(p++);
    pre_b = F(p++); pre_w = F(p++);
    x = F(p++);
  }

  // workspace bump allocator (256B aligned); total ~28MB
  char* wsb = (char*)d_ws;
  size_t off = 0;
  auto alloc = [&](size_t bytes) -> char* {
    char* p = wsb + off;
    off += (bytes + 255) & ~(size_t)255;
    return p;
  };
  float* pix     = (float*)alloc((size_t)NTOK * 256 * 4);
  float* Wc      = (float*)alloc((size_t)96 * 256 * 4);
  float* bc      = (float*)alloc(96 * 4);
  float* tbuf    = (float*)alloc((size_t)NTOK * EDIM * 4);
  float* hbuf    = (float*)alloc((size_t)NTOK * EDIM * 4);
  float* qkvbuf  = (float*)alloc((size_t)NTOK * 288 * 4);
  _Float16* qhp  = (_Float16*)alloc((size_t)NTOK * EDIM * 2 + 256);
  _Float16* khp  = (_Float16*)alloc((size_t)NTOK * EDIM * 2 + 256);
  _Float16* vtp  = (_Float16*)alloc((size_t)NTOK * EDIM * 2 + 256);
  float* attnout = (float*)alloc((size_t)NTOK * EDIM * 4);
  float* mlph    = (float*)alloc((size_t)NTOK * 384 * 4);
  float* feats   = (float*)alloc((size_t)NTOK * 256 * 4);

  auto gemm = [&](const float* A, const float* W, const float* bias,
                  const float* R, float* C, int M, int N, int K, int act) {
    int waves = (M / 32) * (N / 32);
    hipLaunchKernelGGL(k_gemm, dim3(waves / 4), dim3(128), 0, stream,
                       A, W, bias, R, C, M, N, K, act);
  };
  auto ln = [&](const float* X, const float* g, const float* b, float* Y) {
    hipLaunchKernelGGL(k_layernorm, dim3(NTOK / 4), dim3(128), 0, stream,
                       X, g, b, Y, NTOK);
  };

  // Stage A: depth-mean + 2x2 downsample -> pix (4096 x 256)
  hipLaunchKernelGGL(k_mean_downsample, dim3((256 * NTOK) / 256), dim3(256), 0,
                     stream, x, pix);
  // Fold pre_conv + patch embed weights
  hipLaunchKernelGGL(k_wcomb, dim3(96), dim3(256), 0, stream,
                     patch_w, patch_b, pre_w, pre_b, Wc, bc);
  // tokens = LN(pix @ Wc^T + bc)
  gemm(pix, Wc, bc, nullptr, hbuf, NTOK, EDIM, 256, 0);
  ln(hbuf, patch_ng, patch_nb, tbuf);

  for (int b = 0; b < 4; ++b) {
    int nH = nHs[b];
    int dh = EDIM / nH;
    float qscale = 1.0f / sqrtf((float)dh);
    ln(tbuf, blk[b].n1g, blk[b].n1b, hbuf);
    gemm(hbuf, blk[b].qkv_w, blk[b].qkv_b, nullptr, qkvbuf, NTOK, 288, EDIM, 0);
    hipLaunchKernelGGL(k_repack_qkv, dim3((NTOK * 288) / 256), dim3(256), 0,
                       stream, qkvbuf, qhp, khp, vtp, nH, dh, qscale);
    hipLaunchKernelGGL(k_attn, dim3(nH * 64), dim3(128), 0, stream,
                       qhp, khp, vtp, blk[b].rpb, attnout, nH, dh);
    gemm(attnout, blk[b].proj_w, blk[b].proj_b, tbuf, tbuf, NTOK, EDIM, EDIM, 0);
    ln(tbuf, blk[b].n2g, blk[b].n2b, hbuf);
    gemm(hbuf, blk[b].fc1_w, blk[b].fc1_b, nullptr, mlph, NTOK, 384, EDIM, 1);
    gemm(mlph, blk[b].fc2_w, blk[b].fc2_b, tbuf, tbuf, NTOK, EDIM, 384, 0);
  }

  // final norm + map (96 -> 256)
  ln(tbuf, norm_g, norm_b, hbuf);
  gemm(hbuf, map_w, map_b, nullptr, feats, NTOK, 256, EDIM, 0);

  // Stage Z: upsample + sigmoid + gate
  hipLaunchKernelGGL(k_final, dim3((256 * 128 * 128) / 256), dim3(256), 0,
                     stream, x, feats, (float*)d_out);
}
